// DAGLSTM_59871844106677
// MI455X (gfx1250) — matrix operand order repaired and checked by reading, not compiled, not run on hardware
//
#include <hip/hip_runtime.h>
#include <cmath>

// ---------------- problem constants ----------------
constexpr int NB  = 1024;   // batch
constexpr int NN  = 48;     // nodes
constexpr int NIN = 256;    // input size
constexpr int NH  = 256;    // hidden
constexpr int NL  = 2;      // layers
constexpr int NG  = 4 * NH; // 1024 gate width
constexpr size_t BH = (size_t)NB * NH;       // 262144
constexpr size_t SLOT = (size_t)NL * BH;     // per-slot elements (L,B,H)

constexpr int RB    = 16;    // batch rows per block (persistent kernel)
constexpr int TPB   = 128;   // 4 waves per block
constexpr int LDSTR = 264;   // LDS row stride (bf16 elems): 256 + 8 pad -> conflict-free

// ---------------- vector types for WMMA ----------------
typedef __attribute__((ext_vector_type(16))) __bf16 v16bf;
typedef __attribute__((ext_vector_type(8)))  float  v8f;
typedef __attribute__((ext_vector_type(4)))  unsigned int u32x4;

union Frag {
    v16bf v;
    struct Halves { u32x4 lo; u32x4 hi; } q;
};

__device__ __forceinline__ v8f wmma_bf16(v16bf a, v16bf b, v8f c) {
    return __builtin_amdgcn_wmma_f32_16x16x32_bf16(
        false, a, false, b, (short)0, c, false, false);
}

__device__ __forceinline__ unsigned short f2bf(float x) {
    unsigned int u = __float_as_uint(x);
    unsigned int r = u + 0x7FFFu + ((u >> 16) & 1u);   // RNE
    return (unsigned short)(r >> 16);
}
__device__ __forceinline__ float bf2f(unsigned int s) {
    return __uint_as_float(s << 16);
}
__device__ __forceinline__ float sigf(float x) {
    return 1.0f / (1.0f + __expf(-x));
}

// ---------------- setup kernels ----------------
__global__ void f32_to_bf16_k(const float* __restrict__ src,
                              unsigned short* __restrict__ dst, int n) {
    int i = blockIdx.x * blockDim.x + threadIdx.x;
    if (i < n) dst[i] = f2bf(src[i]);
}

__global__ void bias_comb_k(const float* __restrict__ a, const float* __restrict__ b,
                            float* __restrict__ out, int n) {
    int i = blockIdx.x * blockDim.x + threadIdx.x;
    if (i < n) out[i] = a[i] + b[i];
}

// ---------------- parallel hoisted GEMM: XW0 = x @ w_ih0^T + b ----------------
// C[M,1024] f32 = A[M,256]_bf16 @ B[1024,256]_bf16^T + bias
// grid = (1024/128, M/64), 256 threads (8 waves, wave tile 32x32)
__global__ __launch_bounds__(256)
void gemm_wmma_k(const unsigned short* __restrict__ A, int lda,
                 const unsigned short* __restrict__ Bw,
                 const float* __restrict__ bias,
                 float* __restrict__ C) {
    constexpr int KC = 256, LDB = 256, NOUT = 1024;
    const int ln = threadIdx.x & 31;
    const int wv = threadIdx.x >> 5;
    const int wm = wv >> 2, wn = wv & 3;
    const int m0 = blockIdx.y * 64 + wm * 32;
    const int n0 = blockIdx.x * 128 + wn * 32;
    const int lrow = ln & 15;
    const int khA = (ln >> 4) << 3;
    const int khB = (ln >> 4) << 3;
    const int rbase = (ln >> 4) << 3;

    v8f acc[2][2] = {};
    const unsigned short* pa = A + (size_t)(m0 + lrow) * lda + khA;
    const unsigned short* pb = Bw + (size_t)(n0 + lrow) * LDB + khB;
    #pragma unroll
    for (int k = 0; k < KC; k += 32) {
        Frag a0, a1, b0, b1;
        a0.q.lo = *(const u32x4*)(pa + k);
        a0.q.hi = *(const u32x4*)(pa + k + 16);
        a1.q.lo = *(const u32x4*)(pa + k + (size_t)16 * lda);
        a1.q.hi = *(const u32x4*)(pa + k + (size_t)16 * lda + 16);
        b0.q.lo = *(const u32x4*)(pb + k);
        b0.q.hi = *(const u32x4*)(pb + k + 16);
        b1.q.lo = *(const u32x4*)(pb + k + 16 * LDB);
        b1.q.hi = *(const u32x4*)(pb + k + 16 * LDB + 16);
        acc[0][0] = wmma_bf16(a0.v, b0.v, acc[0][0]);
        acc[0][1] = wmma_bf16(a0.v, b1.v, acc[0][1]);
        acc[1][0] = wmma_bf16(a1.v, b0.v, acc[1][0]);
        acc[1][1] = wmma_bf16(a1.v, b1.v, acc[1][1]);
    }
    #pragma unroll
    for (int i = 0; i < 2; ++i) {
        #pragma unroll
        for (int j = 0; j < 2; ++j) {
            const int col = n0 + j * 16 + lrow;
            const float bv = bias[col];
            float* pc = C + (size_t)(m0 + i * 16 + rbase) * NOUT + col;
            #pragma unroll
            for (int r = 0; r < 8; ++r)
                pc[(size_t)r * NOUT] = acc[i][j][r] + bv;
        }
    }
}

// ---------------- persistent per-batch-slice DAG-LSTM kernel ----------------
// One block owns RB=16 batch rows and runs all 48 nodes x 2 layers.
// Wave wv owns hidden units [wv*64, wv*64+64): its 16 C-frags are
// (jt in 0..3) x (gate in 0..3), so i/f/g/o for a (row,hidden) pair sit in the
// same lane across 4 accumulator frags -> elementwise LSTM is register-only.

// A-fragment GEMM helper, A staged in LDS (row stride LDSTR), B = weight [1024,256] bf16.
__device__ __forceinline__ void gemm_tile_lds(v8f acc[4][4],
                                              const unsigned short* sA,
                                              const unsigned short* __restrict__ Bw,
                                              int wv, int lrow, int khA, int khB) {
    const unsigned short* pa = sA + lrow * LDSTR + khA;
    #pragma unroll
    for (int k = 0; k < 256; k += 32) {
        Frag a;
        a.q.lo = *(const u32x4*)(pa + k);
        a.q.hi = *(const u32x4*)(pa + k + 16);
        #pragma unroll
        for (int g = 0; g < 4; ++g) {
            #pragma unroll
            for (int jt = 0; jt < 4; ++jt) {
                const unsigned short* pb =
                    Bw + (size_t)(g * 256 + wv * 64 + jt * 16 + lrow) * 256 + khB + k;
                Frag b; b.q.lo = *(const u32x4*)pb; b.q.hi = *(const u32x4*)(pb + 16);
                acc[jt][g] = wmma_bf16(a.v, b.v, acc[jt][g]);
            }
        }
    }
}

__global__ __launch_bounds__(TPB)
void dag_lstm_persistent_k(const unsigned short* __restrict__ xbf,  // [B,N,256] bf16
                           const float* __restrict__ XW0,           // [B*N,1024] (use_xw0)
                           const unsigned short* __restrict__ wih0,
                           const unsigned short* __restrict__ whh0,
                           const unsigned short* __restrict__ wih1,
                           const unsigned short* __restrict__ whh1,
                           const float* __restrict__ bias0,
                           const float* __restrict__ bias1,
                           const int* __restrict__ pred,
                           unsigned short* __restrict__ Hbuf,       // [(N+1),L,B,H] bf16
                           float* __restrict__ Cbuf,                // [(N+1),L,B,H] f32
                           float* __restrict__ out,                 // [B,H] f32
                           int use_xw0) {
    __shared__ __align__(16) unsigned short sh_hin0[RB * LDSTR];
    __shared__ __align__(16) unsigned short sh_hin1[RB * LDSTR];
    __shared__ __align__(16) unsigned short sh_h0[RB * LDSTR];

    const int tid  = threadIdx.x;
    const int ln   = tid & 31;
    const int wv   = tid >> 5;          // 0..3: owns hidden [wv*64, wv*64+64)
    const int lrow = ln & 15;
    const int khA  = (ln >> 4) << 3;
    const int khB  = (ln >> 4) << 3;
    const int rbase = (ln >> 4) << 3;
    const int row0 = blockIdx.x * RB;   // global batch row base

    for (int node = 0; node < NN; ++node) {
        const int p0 = pred[2 * node + 0];
        const int p1 = pred[2 * node + 1];

        // ---- stage h_in (both layers) into LDS: mean of predecessor slots ----
        for (int e = tid; e < 2 * RB * 128; e += TPB) {
            const int l   = e >> 11;              // RB*128 = 2048
            const int rem = e & 2047;
            const int r   = rem >> 7;             // row in block
            const int kk  = (rem & 127) << 1;     // bf16 pair index
            const size_t ro = (size_t)(row0 + r) * NH + kk;
            const unsigned int a = *(const unsigned int*)
                (Hbuf + ((size_t)p0 * NL + l) * BH + ro);
            const unsigned int b = *(const unsigned int*)
                (Hbuf + ((size_t)p1 * NL + l) * BH + ro);
            const float lo = 0.5f * (bf2f(a & 0xFFFFu) + bf2f(b & 0xFFFFu));
            const float hi = 0.5f * (bf2f(a >> 16)     + bf2f(b >> 16));
            unsigned short* dst = (l ? sh_hin1 : sh_hin0) + r * LDSTR + kk;
            *(unsigned int*)dst = (unsigned int)f2bf(lo) | ((unsigned int)f2bf(hi) << 16);
        }
        __syncthreads();

        // ================= layer 0 =================
        {
            v8f acc[4][4] = {};
            gemm_tile_lds(acc, sh_hin0, whh0, wv, lrow, khA, khB);
            if (!use_xw0) {
                // fold x @ w_ih0^T in directly (A from global, row stride NN*NIN)
                const unsigned short* pa =
                    xbf + ((size_t)(row0 + lrow) * NN + node) * NIN + khA;
                #pragma unroll
                for (int k = 0; k < 256; k += 32) {
                    Frag a;
                    a.q.lo = *(const u32x4*)(pa + k);
                    a.q.hi = *(const u32x4*)(pa + k + 16);
                    #pragma unroll
                    for (int g = 0; g < 4; ++g) {
                        #pragma unroll
                        for (int jt = 0; jt < 4; ++jt) {
                            const unsigned short* pb =
                                wih0 + (size_t)(g * 256 + wv * 64 + jt * 16 + lrow) * 256 + khB + k;
                            Frag b; b.q.lo = *(const u32x4*)pb; b.q.hi = *(const u32x4*)(pb + 16);
                            acc[jt][g] = wmma_bf16(a.v, b.v, acc[jt][g]);
                        }
                    }
                }
            }
            // elementwise (register-local): gates i,f,g,o live in 4 frags, same lane
            #pragma unroll
            for (int jt = 0; jt < 4; ++jt) {
                const int hu = wv * 64 + jt * 16 + lrow;
                #pragma unroll
                for (int r = 0; r < 8; ++r) {
                    const int row = rbase + r;
                    const size_t grow = (size_t)(row0 + row);
                    float gi = acc[jt][0][r], gf = acc[jt][1][r];
                    float gg = acc[jt][2][r], go = acc[jt][3][r];
                    if (use_xw0) {
                        const float* ad = XW0 + (grow * NN + node) * NG + hu;
                        gi += ad[0]; gf += ad[256]; gg += ad[512]; go += ad[768];
                    } else {
                        gi += bias0[hu];       gf += bias0[hu + 256];
                        gg += bias0[hu + 512]; go += bias0[hu + 768];
                    }
                    const size_t eo = grow * NH + hu;
                    const float cin = 0.5f * (Cbuf[((size_t)p0 * NL + 0) * BH + eo] +
                                              Cbuf[((size_t)p1 * NL + 0) * BH + eo]);
                    const float cn = sigf(gf) * cin + sigf(gi) * tanhf(gg);
                    const float hn = sigf(go) * tanhf(cn);
                    Cbuf[((size_t)(node + 1) * NL + 0) * BH + eo] = cn;
                    const unsigned short hb = f2bf(hn);
                    Hbuf[((size_t)(node + 1) * NL + 0) * BH + eo] = hb;
                    sh_h0[row * LDSTR + hu] = hb;
                }
            }
        }
        __syncthreads();   // sh_h0 ready for all waves

        // ================= layer 1 =================
        {
            v8f acc[4][4] = {};
            gemm_tile_lds(acc, sh_h0,   wih1, wv, lrow, khA, khB);
            gemm_tile_lds(acc, sh_hin1, whh1, wv, lrow, khA, khB);
            #pragma unroll
            for (int jt = 0; jt < 4; ++jt) {
                const int hu = wv * 64 + jt * 16 + lrow;
                const float b_i = bias1[hu],       b_f = bias1[hu + 256];
                const float b_g = bias1[hu + 512], b_o = bias1[hu + 768];
                #pragma unroll
                for (int r = 0; r < 8; ++r) {
                    const int row = rbase + r;
                    const size_t grow = (size_t)(row0 + row);
                    const float gi = acc[jt][0][r] + b_i;
                    const float gf = acc[jt][1][r] + b_f;
                    const float gg = acc[jt][2][r] + b_g;
                    const float go = acc[jt][3][r] + b_o;
                    const size_t eo = grow * NH + hu;
                    const float cin = 0.5f * (Cbuf[((size_t)p0 * NL + 1) * BH + eo] +
                                              Cbuf[((size_t)p1 * NL + 1) * BH + eo]);
                    const float cn = sigf(gf) * cin + sigf(gi) * tanhf(gg);
                    const float hn = sigf(go) * tanhf(cn);
                    Cbuf[((size_t)(node + 1) * NL + 1) * BH + eo] = cn;
                    Hbuf[((size_t)(node + 1) * NL + 1) * BH + eo] = f2bf(hn);
                    if (node == NN - 1) out[eo] = hn;
                }
            }
        }
        // make this node's Hbuf/Cbuf writes visible block-wide before next staging
        __threadfence_block();
        __syncthreads();
    }
}

// ---------------- host-side launch ----------------
extern "C" void kernel_launch(void* const* d_in, const int* in_sizes, int n_in,
                              void* d_out, int out_size, void* d_ws, size_t ws_size,
                              hipStream_t stream) {
    const float* dags  = (const float*)d_in[0];
    const float* h0    = (const float*)d_in[1];
    const float* c0    = (const float*)d_in[2];
    const float* w_ih0 = (const float*)d_in[3];
    const float* w_hh0 = (const float*)d_in[4];
    const float* b_ih0 = (const float*)d_in[5];
    const float* b_hh0 = (const float*)d_in[6];
    const float* w_ih1 = (const float*)d_in[7];
    const float* w_hh1 = (const float*)d_in[8];
    const float* b_ih1 = (const float*)d_in[9];
    const float* b_hh1 = (const float*)d_in[10];
    const int*   pred  = (const int*)d_in[11];
    (void)in_sizes; (void)n_in; (void)out_size;

    // ---- carve workspace (256B aligned) ----
    char* base = (char*)d_ws;
    size_t off = 0;
    auto carve = [&](size_t bytes) -> char* {
        char* p = base + off;
        off = (off + bytes + 255) & ~(size_t)255;
        return p;
    };
    constexpr size_t WELEM = (size_t)NG * NIN;
    unsigned short* wih0 = (unsigned short*)carve(WELEM * 2);
    unsigned short* whh0 = (unsigned short*)carve(WELEM * 2);
    unsigned short* wih1 = (unsigned short*)carve(WELEM * 2);
    unsigned short* whh1 = (unsigned short*)carve(WELEM * 2);
    float* bias0 = (float*)carve(NG * 4);
    float* bias1 = (float*)carve(NG * 4);
    unsigned short* xbf  = (unsigned short*)carve((size_t)NB * NN * NIN * 2);
    unsigned short* Hbuf = (unsigned short*)carve((size_t)(NN + 1) * SLOT * 2);
    float* Cbuf          = (float*)carve((size_t)(NN + 1) * SLOT * 4);
    float* XW0           = (float*)carve((size_t)NB * NN * NG * 4);
    const int use_xw0 = (off <= ws_size) ? 1 : 0;

    const int T = 256;
    f32_to_bf16_k<<<(int)((WELEM + T - 1) / T), T, 0, stream>>>(w_ih0, wih0, (int)WELEM);
    f32_to_bf16_k<<<(int)((WELEM + T - 1) / T), T, 0, stream>>>(w_hh0, whh0, (int)WELEM);
    f32_to_bf16_k<<<(int)((WELEM + T - 1) / T), T, 0, stream>>>(w_ih1, wih1, (int)WELEM);
    f32_to_bf16_k<<<(int)((WELEM + T - 1) / T), T, 0, stream>>>(w_hh1, whh1, (int)WELEM);
    bias_comb_k<<<(NG + T - 1) / T, T, 0, stream>>>(b_ih0, b_hh0, bias0, NG);
    bias_comb_k<<<(NG + T - 1) / T, T, 0, stream>>>(b_ih1, b_hh1, bias1, NG);
    {
        const size_t nx = (size_t)NB * NN * NIN;
        f32_to_bf16_k<<<(int)((nx + T - 1) / T), T, 0, stream>>>(dags, xbf, (int)nx);
    }
    // slot 0 = start state
    f32_to_bf16_k<<<(int)((SLOT + T - 1) / T), T, 0, stream>>>(h0, Hbuf, (int)SLOT);
    hipMemcpyAsync(Cbuf, c0, SLOT * sizeof(float), hipMemcpyDeviceToDevice, stream);

    if (use_xw0) {
        dim3 grid(NG / 128, (NB * NN) / 64);
        gemm_wmma_k<<<grid, 256, 0, stream>>>(xbf, NIN, wih0, bias0, XW0);
    }

    dag_lstm_persistent_k<<<NB / RB, TPB, 0, stream>>>(
        xbf, XW0, wih0, whh0, wih1, whh1, bias0, bias1, pred,
        Hbuf, Cbuf, (float*)d_out, use_xw0);
}
